// SegmentationModel_TorchScript_16406775071117
// MI455X (gfx1250) — matrix operand order repaired and checked by reading, not compiled, not run on hardware
//
#include <hip/hip_runtime.h>
#include <hip/hip_bf16.h>
#include <cmath>

typedef __attribute__((ext_vector_type(16))) _Float16 v16h;
typedef __attribute__((ext_vector_type(8)))  float    v8f;

#define PATCH 512
#define SLACK 64
#define HS    32    // SLACK/2
#define STEP  448   // PATCH - SLACK

struct Geom {
    int p0, p1, ph, pw;        // patch origin (y,x) and dims in image coords
    int gi0, gj1, gi2, gj3;    // paste rect in image coords: rows [gi0,gi2), cols [gj1,gj3)
};

__device__ inline Geom compute_geom(int i, int H, int Wd) {
    Geom g;
    int gh = (H - SLACK + STEP - 1) / STEP;
    int gw = (Wd - SLACK + STEP - 1) / STEP;
    int r = i / gw, c = i % gw;
    int gy = min(PATCH + r * STEP, H);
    int gx = min(PATCH + c * STEP, Wd);
    g.p0 = max(gy - PATCH, 0);
    g.p1 = max(gx - PATCH, 0);
    g.ph = gy - g.p0;
    g.pw = gx - g.p1;
    // lh = (grid[gh-2,gw-2,2:4] - grid[gh-1,gw-1,0:2]) // 2
    int gy2 = min(PATCH + (gh - 2) * STEP, H);
    int gx2 = min(PATCH + (gw - 2) * STEP, Wd);
    int gyl = min(PATCH + (gh - 1) * STEP, H);
    int gxl = min(PATCH + (gw - 1) * STEP, Wd);
    int lh0 = (gy2 - max(gyl - PATCH, 0)) / 2;
    int lh1 = (gx2 - max(gxl - PATCH, 0)) / 2;
    int crop_top  = (r == 0) ? 0 : ((r == gh - 1) ? lh0 : HS);
    int crop_left = (c == 0) ? 0 : ((c == gw - 1) ? lh1 : HS);
    g.gi0 = g.p0 + crop_top;
    g.gj1 = g.p1 + crop_left;
    g.gi2 = (r < gh - 1) ? (gy - HS) : H;   // python slice clamps gy+H -> H
    g.gj3 = (c < gw - 1) ? (gx - HS) : Wd;
    return g;
}

// ---------------------------------------------------------------------------
// Kernel 1: background — pure HBM-bandwidth float4 copy of y (or zeros if i==0)
// ---------------------------------------------------------------------------
__global__ __launch_bounds__(256)
void bg_copy(const float4* __restrict__ y, float4* __restrict__ out,
             const int* __restrict__ pi, int n4) {
    int idx = blockIdx.x * blockDim.x + threadIdx.x;
    if (idx >= n4) return;
    if (*pi > 0) {
        out[idx] = y[idx];
    } else {
        out[idx] = make_float4(0.f, 0.f, 0.f, 0.f);
    }
}

// ---------------------------------------------------------------------------
// Kernel 2: conv(3x3, 3->1) + bias + sigmoid over the paste region, via WMMA.
// One wave computes a 1x16 output strip with a single v_wmma_f32_16x16x32_f16:
//   A (16x32 f16): row m = pixel (Y, X0+m), K = 27 conv taps (raw u8, exact)
//   B (32x16 f16): B[k,n] = W[k]/255 for all n  (degenerate N)
//   C (16x16 f32): bias broadcast
// ---------------------------------------------------------------------------
__global__ __launch_bounds__(256)
void conv_paste(const int* __restrict__ x_u8, const float* __restrict__ Wt,
                const float* __restrict__ bias, const int* __restrict__ pi,
                float* __restrict__ out, int H, int Wd) {
    const int i = *pi;
    const Geom g = compute_geom(i, H, Wd);

    const int lane = threadIdx.x & 31;
    const int wave = threadIdx.x >> 5;
    const int wid  = blockIdx.x * 8 + wave;      // 8 waves / block
    const int STRIPS = PATCH / 16;               // max strips per region row (32)
    const int srow = wid / STRIPS;
    const int scol = wid % STRIPS;
    const int Y  = g.gi0 + srow;
    const int X0 = g.gj1 + scol * 16;
    if (Y >= g.gi2 || X0 >= g.gj3) return;       // wave-uniform exit (EXEC stays full)

    // A-matrix K mapping (ISA 7.12.2, 16-bit A 16x32):
    //   lanes 0-15 : elements 0..7 -> K 0..7,  elements 8..15 -> K 16..23
    //   lanes 16-31: elements 0..7 -> K 8..15, elements 8..15 -> K 24..31
    const int M   = lane & 15;
    const int kb0 = (lane < 16) ? 0 : 8;
    const int kb1 = (lane < 16) ? 16 : 24;
    const int py  = Y - g.p0;          // patch row of this strip
    const int px  = X0 + M - g.p1;     // patch col of this lane's pixel

    v16h a;
    v16h bm;
    #pragma unroll
    for (int e = 0; e < 16; ++e) {
        // ---- A element: tap kA for pixel (py, px), raw u8 value (exact f16)
        const int kA = (e < 8) ? (kb0 + e) : (kb1 + (e - 8));
        _Float16 av = (_Float16)0.0f;
        if (kA < 27) {
            const int cch = kA / 9;
            const int t   = kA % 9;
            const int kh  = t / 3;
            const int kw  = t % 3;
            const int ty  = py + kh - 1;
            const int tx  = px + kw - 1;
            if (ty >= 0 && ty < g.ph && tx >= 0 && tx < g.pw) {   // SAME zero-pad at patch edge
                const int v = x_u8[(((g.p0 + ty) * (long)Wd) + (g.p1 + tx)) * 3 + cch];
                av = (_Float16)(float)v;
            }
        }
        a[e] = av;
        // ---- B element: K mapping for 16-bit B 32x16:
        //   lanes 0-15 hold K=0..7,16..23; lanes 16-31 hold K=8..15,24..31 (same as A)
        const int kB = ((e < 8) ? 0 : 8) + ((lane < 16) ? 0 : 8) + e;
        bm[e] = (kB < 27) ? (_Float16)(Wt[kB] * (1.0f / 255.0f)) : (_Float16)0.0f;
    }

    const float bb = bias[0];
    v8f cacc = {bb, bb, bb, bb, bb, bb, bb, bb};
    // D = A x B + C   (f32 accumulate)
    v8f d = __builtin_amdgcn_wmma_f32_16x16x32_f16(
        /*neg_a=*/false, a, /*neg_b=*/false, bm,
        /*c_mod=*/(short)0, cacc, /*reuse_a=*/false, /*reuse_b=*/false);

    // D layout: VGPR r -> M=r (lanes 0-15) / M=r+8 (lanes 16-31); all N identical.
    // lane r writes pixel r; lane 16+r writes pixel 8+r.
    const int lr   = lane & 15;
    const int half = lane >> 4;
    if (lr < 8) {
        const int m = lr + 8 * half;
        const int X = X0 + m;
        if (X < g.gj3) {
            const float v = d[lr];
            out[(long)Y * Wd + X] = 1.0f / (1.0f + expf(-v));
        }
    }
}

extern "C" void kernel_launch(void* const* d_in, const int* in_sizes, int n_in,
                              void* d_out, int out_size, void* d_ws, size_t ws_size,
                              hipStream_t stream) {
    const int*   x_u8 = (const int*)  d_in[0];   // (1,H,W,3) int32 0..255
    const float* y    = (const float*)d_in[1];   // (1,1,H,W) f32
    const float* Wt   = (const float*)d_in[2];   // (1,3,3,3) f32 -> 27 flat
    const float* bias = (const float*)d_in[3];   // (1,) f32
    const int*   pi   = (const int*)  d_in[4];   // scalar patch index i

    float* out = (float*)d_out;

    const int hw = in_sizes[1];                  // H*W
    int H = (int)llround(sqrt((double)hw));
    if (H < 1) H = 1;
    const int Wd = hw / H;

    // 1) stream y -> out (or zeros): bandwidth-bound, ~128 MiB total traffic
    const int n4 = hw / 4;
    const int cblocks = (n4 + 255) / 256;
    bg_copy<<<cblocks, 256, 0, stream>>>((const float4*)y, (float4*)out, pi, n4);

    // 2) WMMA conv + sigmoid over the paste region (<= 512x512 -> 512*32 waves)
    const int waves   = PATCH * (PATCH / 16);    // 16384 strips max
    const int pblocks = waves / 8;               // 8 waves per 256-thread block
    conv_paste<<<pblocks, 256, 0, stream>>>(x_u8, Wt, bias, pi, out, H, Wd);
}
